// GTLayer_49709951484794
// MI455X (gfx1250) — compile-verified, matrix operand order checked by reading, not run on hardware
//
#include <hip/hip_runtime.h>

#define N_NODES 50000
#define N_EDGES 800000
#define D_MODEL 128
#define N_HEAD  8

typedef __attribute__((ext_vector_type(16))) __bf16 v16bf;
typedef __attribute__((ext_vector_type(8)))  float  v8f;

// ---------- helpers ----------

__device__ __forceinline__ unsigned short f32_to_bf16(float f) {
    unsigned int u = __float_as_uint(f);
    unsigned int rnd = 0x7FFFu + ((u >> 16) & 1u);   // round-to-nearest-even
    u += rnd;
    return (unsigned short)(u >> 16);
}

// Load a 16x32 bf16 fragment (A layout) from a row-major [*, 128] bf16 array.
// Lane l (l<16): row = row0+l, K = k0+{0..7, 16..23}
// Lane l (l>=16): row = row0+l-16, K = k0+{8..15, 24..31}
__device__ __forceinline__ v16bf load_frag(const unsigned short* __restrict__ base,
                                           int row0, int k0, int lane) {
    int m    = lane & 15;
    int half = lane >> 4;
    const unsigned short* p = base + (size_t)(row0 + m) * D_MODEL + k0 + half * 8;
    union { uint4 u[2]; v16bf v; } un;
    un.u[0] = *(const uint4*)(p);
    un.u[1] = *(const uint4*)(p + 16);
    return un.v;
}

// ---------- conversion kernels ----------

__global__ void cvt_embeds_bf16(const float* __restrict__ x,
                                unsigned short* __restrict__ y, int n) {
    int i = blockIdx.x * blockDim.x + threadIdx.x;
    if (i < n) y[i] = f32_to_bf16(x[i]);
}

// W is [k][n] row-major fp32; write Wt as [n][k] bf16 so B-fragments load like A.
__global__ void cvt_w_transpose(const float* __restrict__ Wq,
                                const float* __restrict__ Wk,
                                const float* __restrict__ Wv,
                                unsigned short* __restrict__ Wqt,
                                unsigned short* __restrict__ Wkt,
                                unsigned short* __restrict__ Wvt) {
    int i = blockIdx.x * blockDim.x + threadIdx.x;   // 0 .. 16383
    if (i >= D_MODEL * D_MODEL) return;
    int k = i >> 7, n = i & 127;
    int o = n * D_MODEL + k;
    Wqt[o] = f32_to_bf16(Wq[i]);
    Wkt[o] = f32_to_bf16(Wk[i]);
    Wvt[o] = f32_to_bf16(Wv[i]);
}

// ---------- fused Q/K/V projection via WMMA ----------
// Block = 256 threads = 8 waves. Block owns a 16-row node tile; wave w owns
// output columns [16w, 16w+16). Each wave: 4 K-steps x 3 WMMAs, A shared.

__global__ __launch_bounds__(256) void qkv_gemm(
    const unsigned short* __restrict__ Ebf,
    const unsigned short* __restrict__ Wqt,
    const unsigned short* __restrict__ Wkt,
    const unsigned short* __restrict__ Wvt,
    float* __restrict__ Q, float* __restrict__ K, float* __restrict__ V) {
    int lane = threadIdx.x & 31;
    int wave = threadIdx.x >> 5;          // 0..7 -> N tile
    int row0 = blockIdx.x * 16;           // 3125 blocks * 16 = 50000 exactly
    int n0   = wave * 16;

    v8f cq = {}, ck = {}, cv = {};
#pragma unroll
    for (int k0 = 0; k0 < D_MODEL; k0 += 32) {
        v16bf a  = load_frag(Ebf, row0, k0, lane);
        v16bf bq = load_frag(Wqt, n0,  k0, lane);
        v16bf bk = load_frag(Wkt, n0,  k0, lane);
        v16bf bv = load_frag(Wvt, n0,  k0, lane);
        cq = __builtin_amdgcn_wmma_f32_16x16x32_bf16(false, a, false, bq, (short)0, cq, false, false);
        ck = __builtin_amdgcn_wmma_f32_16x16x32_bf16(false, a, false, bk, (short)0, ck, false, false);
        cv = __builtin_amdgcn_wmma_f32_16x16x32_bf16(false, a, false, bv, (short)0, cv, false, false);
    }

    // C/D layout: VGPR r, lanes 0-15 -> (M=r, N=lane); lanes 16-31 -> (M=r+8, N=lane-16)
    int col   = n0 + (lane & 15);
    int rbase = row0 + ((lane >> 4) << 3);
#pragma unroll
    for (int r = 0; r < 8; ++r) {
        size_t o = (size_t)(rbase + r) * D_MODEL + col;
        Q[o] = cq[r]; K[o] = ck[r]; V[o] = cv[r];
    }
}

// ---------- edge pass 1: exp(clip(q.k)) + denominator scatter ----------

__global__ __launch_bounds__(256) void attn_edge(
    const int* __restrict__ rows, const int* __restrict__ cols,
    const float* __restrict__ Q, const float* __restrict__ K,
    float* __restrict__ expAtt, float* __restrict__ attNorm) {
    int idx = blockIdx.x * blockDim.x + threadIdx.x;
    if (idx >= N_EDGES * N_HEAD) return;
    int e = idx >> 3, h = idx & 7;
    int r = rows[e], c = cols[e];
    const float4* q = (const float4*)(Q + (size_t)r * D_MODEL + h * 16);
    const float4* k = (const float4*)(K + (size_t)c * D_MODEL + h * 16);
    float acc = 0.f;
#pragma unroll
    for (int i = 0; i < 4; ++i) {
        float4 a = q[i], b = k[i];
        acc += a.x * b.x + a.y * b.y + a.z * b.z + a.w * b.w;
    }
    acc = fminf(10.f, fmaxf(-10.f, acc));
    float ex = __expf(acc);
    expAtt[idx] = ex;
    atomicAdd(&attNorm[(size_t)r * N_HEAD + h], ex);
}

// ---------- edge pass 2: normalize + scatter att*v ----------
// 32 threads per edge; thread j handles 4 contiguous cols (head h = j>>2).

__global__ __launch_bounds__(256) void scatter_v(
    const int* __restrict__ rows, const int* __restrict__ cols,
    const float* __restrict__ V, const float* __restrict__ expAtt,
    const float* __restrict__ attNorm, float* __restrict__ accum) {
    long long idx = (long long)blockIdx.x * blockDim.x + threadIdx.x;
    if (idx >= (long long)N_EDGES * 32) return;
    int e = (int)(idx >> 5);
    int j = (int)(idx & 31);
    int h = j >> 2;
    int r = rows[e], c = cols[e];
    float w = expAtt[(size_t)e * N_HEAD + h] /
              (attNorm[(size_t)r * N_HEAD + h] + 1e-8f);
    float4 v = *(const float4*)(V + (size_t)c * D_MODEL + j * 4);
    float* dst = accum + (size_t)r * D_MODEL + j * 4;
    atomicAdd(dst + 0, w * v.x);
    atomicAdd(dst + 1, w * v.y);
    atomicAdd(dst + 2, w * v.z);
    atomicAdd(dst + 3, w * v.w);
}

// ---------- residual + LayerNorm: one wave per node ----------

__global__ __launch_bounds__(256) void layernorm_out(
    const float* __restrict__ accum, const float* __restrict__ emb,
    const float* __restrict__ gamma, const float* __restrict__ beta,
    float* __restrict__ out) {
    int node = blockIdx.x * 8 + (threadIdx.x >> 5);   // 6250 blocks * 8 = 50000
    int lane = threadIdx.x & 31;
    size_t base = (size_t)node * D_MODEL + lane * 4;

    float4 a = *(const float4*)(accum + base);
    float4 x = *(const float4*)(emb + base);
    float r0 = a.x + x.x, r1 = a.y + x.y, r2 = a.z + x.z, r3 = a.w + x.w;

    float s = r0 + r1 + r2 + r3;
#pragma unroll
    for (int off = 16; off > 0; off >>= 1) s += __shfl_xor(s, off, 32);
    float mean = s * (1.f / D_MODEL);

    float d0 = r0 - mean, d1 = r1 - mean, d2 = r2 - mean, d3 = r3 - mean;
    float vs = d0 * d0 + d1 * d1 + d2 * d2 + d3 * d3;
#pragma unroll
    for (int off = 16; off > 0; off >>= 1) vs += __shfl_xor(vs, off, 32);
    float inv = rsqrtf(vs * (1.f / D_MODEL) + 1e-6f);

    float4 g = *(const float4*)(gamma + lane * 4);
    float4 b = *(const float4*)(beta + lane * 4);
    float4 o;
    o.x = d0 * inv * g.x + b.x;
    o.y = d1 * inv * g.y + b.y;
    o.z = d2 * inv * g.z + b.z;
    o.w = d3 * inv * g.w + b.w;
    *(float4*)(out + base) = o;
}

// ---------- launch ----------

extern "C" void kernel_launch(void* const* d_in, const int* in_sizes, int n_in,
                              void* d_out, int out_size, void* d_ws, size_t ws_size,
                              hipStream_t stream) {
    const int*   edge   = (const int*)d_in[0];       // [2, E] flat
    const int*   rows   = edge;
    const int*   cols   = edge + N_EDGES;
    const float* embeds = (const float*)d_in[1];
    const float* Wq     = (const float*)d_in[2];
    const float* Wk     = (const float*)d_in[3];
    const float* Wv     = (const float*)d_in[4];
    const float* gamma  = (const float*)d_in[5];
    const float* beta   = (const float*)d_in[6];
    float*       out    = (float*)d_out;

    char* ws = (char*)d_ws;
    size_t off = 0;
    auto carve = [&](size_t bytes) {
        void* p = ws + off;
        off += (bytes + 255) & ~(size_t)255;
        return p;
    };
    unsigned short* Ebf  = (unsigned short*)carve((size_t)N_NODES * D_MODEL * 2);
    unsigned short* Wqt  = (unsigned short*)carve((size_t)D_MODEL * D_MODEL * 2);
    unsigned short* Wkt  = (unsigned short*)carve((size_t)D_MODEL * D_MODEL * 2);
    unsigned short* Wvt  = (unsigned short*)carve((size_t)D_MODEL * D_MODEL * 2);
    float* Q       = (float*)carve((size_t)N_NODES * D_MODEL * 4);
    float* K       = (float*)carve((size_t)N_NODES * D_MODEL * 4);
    float* V       = (float*)carve((size_t)N_NODES * D_MODEL * 4);
    float* expAtt  = (float*)carve((size_t)N_EDGES * N_HEAD * 4);
    float* attNorm = (float*)carve((size_t)N_NODES * N_HEAD * 4);
    float* accum   = (float*)carve((size_t)N_NODES * D_MODEL * 4);

    // zero the scatter targets (graph-capture-safe)
    hipMemsetAsync(attNorm, 0, (size_t)N_NODES * N_HEAD * 4, stream);
    hipMemsetAsync(accum,   0, (size_t)N_NODES * D_MODEL * 4, stream);

    int nE = N_NODES * D_MODEL;
    cvt_embeds_bf16<<<(nE + 255) / 256, 256, 0, stream>>>(embeds, Ebf, nE);
    cvt_w_transpose<<<(D_MODEL * D_MODEL + 255) / 256, 256, 0, stream>>>(
        Wq, Wk, Wv, Wqt, Wkt, Wvt);

    qkv_gemm<<<N_NODES / 16, 256, 0, stream>>>(Ebf, Wqt, Wkt, Wvt, Q, K, V);

    attn_edge<<<(N_EDGES * N_HEAD + 255) / 256, 256, 0, stream>>>(
        rows, cols, Q, K, expAtt, attNorm);

    long long nScat = (long long)N_EDGES * 32;
    scatter_v<<<(unsigned)((nScat + 255) / 256), 256, 0, stream>>>(
        rows, cols, V, expAtt, attNorm, accum);

    layernorm_out<<<N_NODES / 8, 256, 0, stream>>>(accum, embeds, gamma, beta, out);
}
